// Glm5Attention_39642548142356
// MI455X (gfx1250) — matrix operand order repaired and checked by reading, not compiled, not run on hardware
//
#include <hip/hip_runtime.h>

// ---------------------------------------------------------------------------
// MLA attention prefill for gfx1250 (MI455X). bf16 WMMA everywhere.
// T=2048, HID=4096, H=32, DQ=256 (192 nope + 64 rope), DV=256, DKV=512,
// QLORA=2048.  SCALE = 256^-0.5 = 0.0625.
// Tensor Data Mover (tensor_load_to_lds + s_wait_tensorcnt) stages:
//   - the K tile in the attention kernel (32x256 bf16, padded LDS pitch)
//   - the A tile in the bf16-A GEMMs (128x32 bf16), overlapped with the
//     VALU-heavy f32->bf16 B-tile conversion done by all 8 waves.
// ---------------------------------------------------------------------------

typedef __attribute__((ext_vector_type(16))) __bf16    v16bf;
typedef __attribute__((ext_vector_type(8)))  float     v8f;
typedef __attribute__((ext_vector_type(4)))  unsigned  v4u;
typedef __attribute__((ext_vector_type(8)))  unsigned  v8u;

#if defined(__gfx1250__) && __has_builtin(__builtin_amdgcn_s_wait_tensorcnt)
#define USE_TDM 1
#else
#define USE_TDM 0
#endif

union Frag {
    v16bf v;
    unsigned u[8];
};

// A-fragment K index for element pair held in "vgpr" v, lane-half hi.
// (16-bit A 16x32: lanes 0-15 hold K {0..7,16..23}, lanes 16-31 +8)
__device__ __forceinline__ int akidx(int v, int hi) {
    return ((v < 4) ? (2 * v) : (16 + 2 * (v - 4))) + 8 * hi;
}
// B-fragment K index (16-bit B 32x16: lanes 0-15 hold K {0..7,16..23}, lanes 16-31 +8)
__device__ __forceinline__ int bkidx(int v, int hi) {
    return ((v < 4) ? (2 * v) : (16 + 2 * (v - 4))) + 8 * hi;
}

__device__ __forceinline__ v8f wmma_bf16(const Frag& a, const Frag& b, v8f c) {
    return __builtin_amdgcn_wmma_f32_16x16x32_bf16(
        /*neg_a=*/false, a.v, /*neg_b=*/false, b.v,
        /*c_mod=*/(short)0, c, /*reuse_a=*/false, /*reuse_b=*/false);
}

// --- staging helpers: 16 contiguous elements -> bf16 -------------------------
__device__ __forceinline__ void cvt16(const float* p, __bf16* o) {
#pragma unroll
    for (int i = 0; i < 16; i += 4) {
        float4 f = *(const float4*)(p + i);
        o[i + 0] = (__bf16)f.x;
        o[i + 1] = (__bf16)f.y;
        o[i + 2] = (__bf16)f.z;
        o[i + 3] = (__bf16)f.w;
    }
}
__device__ __forceinline__ void cvt16(const __bf16* p, __bf16* o) {
#pragma unroll
    for (int i = 0; i < 16; i += 4)
        *(uint2*)(o + i) = *(const uint2*)(p + i);
}
__device__ __forceinline__ void store_cvt(float* p, float v)  { *p = v; }
__device__ __forceinline__ void store_cvt(__bf16* p, float v) { *p = (__bf16)v; }

// ---------------------------------------------------------------------------
// TDM: DMA a 2D bf16 tile (rows x cols, global row stride in elements) into
// LDS at lds_byte_off with an LDS row pitch padded per the D# pad fields.
// Caller must be wave-uniform; completion via s_wait_tensorcnt.
// Descriptor layout per CDNA5 ISA 8.3/8.4 (2D -> groups 0 and 1 only).
// ---------------------------------------------------------------------------
#if USE_TDM
__device__ __forceinline__ void tdm_load_tile_bf16(
    unsigned lds_byte_off, const void* gsrc, unsigned cols, unsigned rows,
    unsigned row_stride_elems, unsigned pad_interval_code, unsigned pad_amount_code) {
    unsigned long long ga = (unsigned long long)(size_t)gsrc;
    v4u g0;
    g0[0] = 1u;                                   // count = 1 (valid user D#)
    g0[1] = lds_byte_off;                         // lds_addr
    g0[2] = (unsigned)ga;                         // global_addr[31:0]
    g0[3] = (unsigned)((ga >> 32) & 0x1FFFFFFull) // global_addr[56:32]
          | (2u << 30);                           // type = 2 ("image")
    v8u g1;
    g1[0] = (1u << 16)                            // data_size = 2 bytes
          | (1u << 20)                            // pad_enable
          | (pad_interval_code << 22)             // DWORDs between pads
          | (pad_amount_code << 25);              // pad DWORDs - 1
    g1[1] = (cols & 0xFFFFu) << 16;               // tensor_dim0[15:0]
    g1[2] = (cols >> 16) | ((rows & 0xFFFFu) << 16); // dim0 hi | tensor_dim1 lo
    g1[3] = (rows >> 16) | ((cols & 0xFFFFu) << 16); // dim1 hi | tile_dim0
    g1[4] = rows & 0xFFFFu;                       // tile_dim1 (tile_dim2 = 0)
    g1[5] = row_stride_elems;                     // tensor_dim0_stride[31:0]
    g1[6] = 0u;                                   // stride hi, dim1_stride lo
    g1[7] = 0u;
    asm volatile("tensor_load_to_lds %0, %1" :: "s"(g0), "s"(g1) : "memory");
}
#endif

// ---------------------------------------------------------------------------
// Generic C[M,N] = A[M,K] @ B[K,N].  A: fp32 or bf16 (row major),
// B: fp32 weights (row major, converted to bf16 while staging),
// C: fp32 or bf16.  Block: 256 thr (8 waves), 128x128 tile, K step 32.
// Wave wv: rows (wv>>1)*32 .. +31, cols (wv&1)*64 .. +63  -> 2x4 WMMA tiles.
// When A is bf16 and TDM is available, the A tile is staged by the Tensor
// Data Mover (issued before B conversion, waited after -> latency hidden).
// M, K must be multiples of 128/32 (true for all call sites); N is guarded.
// ---------------------------------------------------------------------------
template <typename TA, typename TC>
__global__ __launch_bounds__(256)
void gemm_kernel(const TA* __restrict__ A, const float* __restrict__ B,
                 TC* __restrict__ C, int M, int N, int K) {
    __shared__ __bf16 Asub[128 * 36];   // [row][k], pitch 36 (64B row + 8B pad)
    __shared__ __bf16 Bsub[128 * 36];   // [col][k] (transposed), pitch 36

    const int tid  = threadIdx.x;
    const int m0   = blockIdx.y * 128;
    const int n0   = blockIdx.x * 128;
    const int lane = tid & 31;
    const int wv   = tid >> 5;
    const int hi   = lane >> 4;
    const int r    = lane & 15;
    const int wm   = (wv >> 1) * 32;
    const int wn   = (wv & 1) * 64;

    v8f zero = 0.0f;
    v8f acc[2][4];
#pragma unroll
    for (int i = 0; i < 2; i++)
#pragma unroll
        for (int j = 0; j < 4; j++) acc[i][j] = zero;

    const int ar = tid >> 1;          // A stage: row 0..127
    const int ac = (tid & 1) * 16;    //          k  0 or 16
    const int bk = tid >> 3;          // B stage: k   0..31
    const int bn = (tid & 7) * 16;    //          n   0..112

    for (int k0 = 0; k0 < K; k0 += 32) {
        __syncthreads();
#if USE_TDM
        if constexpr (sizeof(TA) == 2) {
            // A already bf16: 128x32 tile via TDM, LDS pitch 36 elems
            // (pad_interval=3 -> 16 DWORDs/row, pad_amount=1 -> 2 DWORDs).
            if (tid < 32)
                tdm_load_tile_bf16((unsigned)(size_t)(void*)Asub,
                                   A + (size_t)m0 * K + k0,
                                   /*cols=*/32, /*rows=*/128,
                                   /*row_stride=*/(unsigned)K,
                                   /*pad_interval=*/3u, /*pad_amount=*/1u);
        } else {
            const TA* ap = A + (size_t)(m0 + ar) * K + (k0 + ac);
            cvt16(ap, &Asub[ar * 36 + ac]);
        }
#else
        {   // stage A (convert to bf16 if needed)
            const TA* ap = A + (size_t)(m0 + ar) * K + (k0 + ac);
            cvt16(ap, &Asub[ar * 36 + ac]);
        }
#endif
        {   // stage B transposed: Bsub[n][k]  (overlaps the A-tile DMA)
            const float* bp = B + (size_t)(k0 + bk) * N + n0 + bn;
#pragma unroll
            for (int i = 0; i < 16; i++) {
                int col = n0 + bn + i;
                float vv = (col < N) ? bp[i] : 0.0f;
                Bsub[(bn + i) * 36 + bk] = (__bf16)vv;
            }
        }
#if USE_TDM
        if constexpr (sizeof(TA) == 2) {
            if (tid < 32) __builtin_amdgcn_s_wait_tensorcnt(0);
        }
#endif
        __syncthreads();

        Frag a[2], b[4];
#pragma unroll
        for (int mt = 0; mt < 2; mt++) {
            int row = wm + mt * 16 + r;
#pragma unroll
            for (int v = 0; v < 8; v++)
                a[mt].u[v] = *(const unsigned*)&Asub[row * 36 + akidx(v, hi)];
        }
#pragma unroll
        for (int nt = 0; nt < 4; nt++) {
            int col = wn + nt * 16 + r;
#pragma unroll
            for (int v = 0; v < 8; v++)
                b[nt].u[v] = *(const unsigned*)&Bsub[col * 36 + bkidx(v, hi)];
        }
#pragma unroll
        for (int mt = 0; mt < 2; mt++)
#pragma unroll
            for (int nt = 0; nt < 4; nt++)
                acc[mt][nt] = wmma_bf16(a[mt], b[nt], acc[mt][nt]);
    }

#pragma unroll
    for (int mt = 0; mt < 2; mt++)
#pragma unroll
        for (int nt = 0; nt < 4; nt++) {
            int col = n0 + wn + nt * 16 + r;
            if (col < N) {
#pragma unroll
                for (int v = 0; v < 8; v++) {
                    int row = m0 + wm + mt * 16 + hi * 8 + v;
                    store_cvt(&C[(size_t)row * N + col], acc[mt][nt][v]);
                }
            }
        }
}

// ---------------------------------------------------------------------------
// RMSNorm one row per block; fp32 in -> bf16 out.
// ---------------------------------------------------------------------------
__global__ __launch_bounds__(256)
void rmsnorm_kernel(const float* __restrict__ x, int xs, int D,
                    const float* __restrict__ w, __bf16* __restrict__ y, int ys) {
    __shared__ float red[256];
    const int row = blockIdx.x;
    const int tid = threadIdx.x;
    const float* xr = x + (size_t)row * xs;
    float ss = 0.0f;
    for (int d = tid; d < D; d += 256) { float v = xr[d]; ss += v * v; }
    red[tid] = ss;
    __syncthreads();
    for (int s = 128; s > 0; s >>= 1) {
        if (tid < s) red[tid] += red[tid + s];
        __syncthreads();
    }
    float inv = rsqrtf(red[0] / (float)D + 1e-6f);
    __bf16* yr = y + (size_t)row * ys;
    for (int d = tid; d < D; d += 256)
        yr[d] = (__bf16)(xr[d] * inv * w[d]);
}

// ---------------------------------------------------------------------------
// RoPE q_pe in place (bf16) + assemble K[t,h,256] = [k_nope(192) | rope(k_pe)].
// latent fp32 [T,576]; k_pe = cols 512..575 (shared across heads).
// ---------------------------------------------------------------------------
__global__ __launch_bounds__(256)
void build_qk_kernel(const float* __restrict__ latent, const int* __restrict__ pos,
                     __bf16* __restrict__ q, const __bf16* __restrict__ kv,
                     __bf16* __restrict__ Kb) {
    __shared__ float cs[32], sn[32], kr[64];
    const int t = blockIdx.x;
    const int tid = threadIdx.x;
    if (tid < 32) {
        float fr  = __powf(10000.0f, -(float)tid / 32.0f);
        float ang = (float)pos[t] * fr;
        float c = __cosf(ang), s = __sinf(ang);
        cs[tid] = c; sn[tid] = s;
        float x1 = latent[(size_t)t * 576 + 512 + tid];
        float x2 = latent[(size_t)t * 576 + 544 + tid];
        kr[tid]      = x1 * c - x2 * s;
        kr[tid + 32] = x2 * c + x1 * s;
    }
    __syncthreads();
    // assemble K row for this token (H*256 = 8192 bf16)
    for (int e = tid; e < 8192; e += 256) {
        int hh = e >> 8, d = e & 255;
        __bf16 val = (d < 192) ? kv[(size_t)t * 14336 + hh * 448 + d]
                               : (__bf16)kr[d - 192];
        Kb[(size_t)t * 8192 + e] = val;
    }
    // rope q_pe in place: per head, pairs (192+i, 224+i)
    for (int p = tid; p < 1024; p += 256) {
        int hh = p >> 5, i = p & 31;
        size_t base = (size_t)t * 8192 + hh * 256 + 192;
        float x1 = (float)q[base + i];
        float x2 = (float)q[base + 32 + i];
        q[base + i]      = (__bf16)(x1 * cs[i] - x2 * sn[i]);
        q[base + 32 + i] = (__bf16)(x2 * cs[i] + x1 * sn[i]);
    }
}

// ---------------------------------------------------------------------------
// Flash attention, causal.  Block = (64 q rows, 1 head), 128 thr = 4 waves,
// each wave owns 16 q rows.  Key tiles of 32 staged in LDS (shared by waves);
// K tile staged by the Tensor Data Mover when available.
// Q: [T, H*256] bf16 (roped)   Kb: [T, H*256] bf16
// V inside KV: [T, H*448] bf16 at col h*448+192..+447
// O: [T, H*256] bf16
// ---------------------------------------------------------------------------
__global__ __launch_bounds__(128)
void mla_attn_kernel(const __bf16* __restrict__ Q, const __bf16* __restrict__ Kb,
                     const __bf16* __restrict__ KV, __bf16* __restrict__ O) {
    __shared__ __bf16 Kl[32 * 264];     // [key][d], pitch 264 (TDM pad: 512B+16B)
    __shared__ __bf16 Vl[256 * 40];     // [d][key] (transposed), pitch 40
    __shared__ __bf16 Pl[4 * 16 * 32];  // per-wave P tile [row][key]

    const int tid  = threadIdx.x;
    const int qt0  = blockIdx.x * 64;
    const int h    = blockIdx.y;
    const int lane = tid & 31;
    const int wv   = tid >> 5;
    const int hi   = lane >> 4;
    const int r    = lane & 15;
    const int rowtok0 = qt0 + wv * 16 + hi * 8;

    // preload Q as 8 A-fragments covering d = 0..255
    Frag aq[8];
    {
        int tok = qt0 + wv * 16 + r;
        const __bf16* qp = Q + (size_t)tok * 8192 + h * 256;
#pragma unroll
        for (int f = 0; f < 8; f++)
#pragma unroll
            for (int v = 0; v < 8; v++)
                aq[f].u[v] = *(const unsigned*)&qp[f * 32 + akidx(v, hi)];
    }

    v8f zero = 0.0f;
    v8f o[16];
#pragma unroll
    for (int t = 0; t < 16; t++) o[t] = zero;
    float m[8], l[8];
#pragma unroll
    for (int v = 0; v < 8; v++) { m[v] = -1e30f; l[v] = 0.0f; }

    const int ntiles = (qt0 >> 5) + 2;   // keys 0 .. qt0+63
    for (int kt = 0; kt < ntiles; kt++) {
        const int kbase = kt * 32;
        __syncthreads();
#if USE_TDM
        // stage K tile [32 keys][256 d] via Tensor Data Mover (wave 0 only;
        // EXEC is ignored by TDM, branch is wave-uniform).  LDS pitch 264
        // elements = 512B row + 16B pad (pad_interval=6 -> 128 DWORDs,
        // pad_amount=3 -> 4 DWORDs).
        if (tid < 32) {
            tdm_load_tile_bf16((unsigned)(size_t)(void*)Kl,
                               Kb + (size_t)kbase * 8192 + h * 256,
                               /*cols=*/256, /*rows=*/32,
                               /*row_stride=*/8192,
                               /*pad_interval=*/6u, /*pad_amount=*/3u);
        }
#else
        // stage K tile [32 keys][256 d]
#pragma unroll
        for (int j = 0; j < 8; j++) {
            int idx = j * 128 + tid;
            int row = idx >> 5;
            int d   = (idx & 31) * 8;
            *(uint4*)&Kl[row * 264 + d] =
                *(const uint4*)&Kb[(size_t)(kbase + row) * 8192 + h * 256 + d];
        }
#endif
        // stage V tile transposed: Vl[d][key]  (overlaps the K-tile DMA)
#pragma unroll
        for (int j = 0; j < 8; j++) {
            int idx = j * 128 + tid;
            int row = idx >> 5;
            int d   = (idx & 31) * 8;
            uint4 val = *(const uint4*)&KV[(size_t)(kbase + row) * 14336 +
                                           h * 448 + 192 + d];
            const __bf16* pv = (const __bf16*)&val;
#pragma unroll
            for (int e = 0; e < 8; e++) Vl[(d + e) * 40 + row] = pv[e];
        }
#if USE_TDM
        if (tid < 32) __builtin_amdgcn_s_wait_tensorcnt(0);
#endif
        __syncthreads();

        // S^ = Q . K^T : two 16x16 tiles (keys 0-15, 16-31)
        v8f s0 = zero, s1 = zero;
#pragma unroll
        for (int f = 0; f < 8; f++) {
            Frag b0, b1;
#pragma unroll
            for (int v = 0; v < 8; v++) {
                int k = f * 32 + bkidx(v, hi);
                b0.u[v] = *(const unsigned*)&Kl[r * 264 + k];
                b1.u[v] = *(const unsigned*)&Kl[(r + 16) * 264 + k];
            }
            s0 = wmma_bf16(aq[f], b0, s0);
            s1 = wmma_bf16(aq[f], b1, s1);
        }

        // online softmax (row stats: butterfly over 16-lane group)
        float p0[8], p1[8], mx[8];
#pragma unroll
        for (int v = 0; v < 8; v++) {
            float a0 = s0[v] * 0.0625f, a1 = s1[v] * 0.0625f;
            int rt = rowtok0 + v;
            if (kbase + r > rt)      a0 = -1e30f;
            if (kbase + r + 16 > rt) a1 = -1e30f;
            p0[v] = a0; p1[v] = a1;
            mx[v] = fmaxf(a0, a1);
        }
#pragma unroll
        for (int off = 1; off < 16; off <<= 1)
#pragma unroll
            for (int v = 0; v < 8; v++)
                mx[v] = fmaxf(mx[v], __shfl_xor(mx[v], off, 32));
        float alpha[8], rs[8];
#pragma unroll
        for (int v = 0; v < 8; v++) {
            float mn = fmaxf(m[v], mx[v]);
            alpha[v] = __expf(m[v] - mn);
            m[v] = mn;
            p0[v] = __expf(p0[v] - mn);
            p1[v] = __expf(p1[v] - mn);
            rs[v] = p0[v] + p1[v];
        }
#pragma unroll
        for (int off = 1; off < 16; off <<= 1)
#pragma unroll
            for (int v = 0; v < 8; v++)
                rs[v] += __shfl_xor(rs[v], off, 32);
#pragma unroll
        for (int v = 0; v < 8; v++) l[v] = l[v] * alpha[v] + rs[v];
#pragma unroll
        for (int t = 0; t < 16; t++)
#pragma unroll
            for (int v = 0; v < 8; v++) o[t][v] *= alpha[v];

        // write P (C layout -> row-major LDS) for transpose into A fragment
        __bf16* pw = &Pl[wv * 512];
#pragma unroll
        for (int v = 0; v < 8; v++) {
            pw[(hi * 8 + v) * 32 + r]      = (__bf16)p0[v];
            pw[(hi * 8 + v) * 32 + r + 16] = (__bf16)p1[v];
        }
        __syncthreads();

        // O += P . V  (16 WMMAs over d = 0..255)
        Frag pa;
#pragma unroll
        for (int v = 0; v < 8; v++)
            pa.u[v] = *(const unsigned*)&pw[r * 32 + akidx(v, hi)];
#pragma unroll
        for (int t = 0; t < 16; t++) {
            Frag bv;
#pragma unroll
            for (int v = 0; v < 8; v++)
                bv.u[v] = *(const unsigned*)&Vl[(t * 16 + r) * 40 + bkidx(v, hi)];
            o[t] = wmma_bf16(pa, bv, o[t]);
        }
    }

    // normalize and store
    float inv[8];
#pragma unroll
    for (int v = 0; v < 8; v++) inv[v] = 1.0f / l[v];
#pragma unroll
    for (int t = 0; t < 16; t++) {
        int col = h * 256 + t * 16 + r;
#pragma unroll
        for (int v = 0; v < 8; v++) {
            int tok = qt0 + wv * 16 + hi * 8 + v;
            O[(size_t)tok * 8192 + col] = (__bf16)(o[t][v] * inv[v]);
        }
    }
}

// ---------------------------------------------------------------------------
extern "C" void kernel_launch(void* const* d_in, const int* in_sizes, int n_in,
                              void* d_out, int out_size, void* d_ws, size_t ws_size,
                              hipStream_t stream) {
    (void)in_sizes; (void)n_in; (void)out_size; (void)ws_size;
    const float* hidden    = (const float*)d_in[0];
    const int*   positions = (const int*)d_in[1];
    const float* wq_a      = (const float*)d_in[2];
    const float* q_a_ln_w  = (const float*)d_in[3];
    const float* wq_b      = (const float*)d_in[4];
    const float* wkv_a     = (const float*)d_in[5];
    const float* kv_a_ln_w = (const float*)d_in[6];
    const float* wkv_b     = (const float*)d_in[7];
    const float* wo        = (const float*)d_in[8];
    float* out = (float*)d_out;

    // workspace layout (bytes); H-region overlays temps then attn_out
    char* ws = (char*)d_ws;
    __bf16* q      = (__bf16*)(ws);                  // 2048*8192*2  = 33554432
    __bf16* kv     = (__bf16*)(ws + 33554432);       // 2048*14336*2 = 58720256
    __bf16* Kb     = (__bf16*)(ws + 92274688);       // 2048*8192*2  = 33554432
    char*   Hb     = ws + 125829120;                 // 33554432 overlay region
    float*  q_lora   = (float*)(Hb);                 // 2048*2048*4 (dead after q)
    __bf16* q_lora_n = (__bf16*)(Hb + 16777216);     // 2048*2048*2
    float*  latent   = (float*)(Hb + 25165824);      // 2048*576*4 (dead after build_qk)
    __bf16* kv_a_n   = (__bf16*)(Hb + 29884416);     // 2048*512*2 (dead after kv gemm)
    __bf16* attn_o   = (__bf16*)Hb;                  // written last, overlays the above

    dim3 blk(256);
    // q path
    gemm_kernel<float, float><<<dim3(16, 16), blk, 0, stream>>>(
        hidden, wq_a, q_lora, 2048, 2048, 4096);
    rmsnorm_kernel<<<2048, 256, 0, stream>>>(q_lora, 2048, 2048, q_a_ln_w,
                                             q_lora_n, 2048);
    gemm_kernel<__bf16, __bf16><<<dim3(64, 16), blk, 0, stream>>>(
        q_lora_n, wq_b, q, 2048, 8192, 2048);
    // kv path
    gemm_kernel<float, float><<<dim3(5, 16), blk, 0, stream>>>(
        hidden, wkv_a, latent, 2048, 576, 4096);
    rmsnorm_kernel<<<2048, 256, 0, stream>>>(latent, 576, 512, kv_a_ln_w,
                                             kv_a_n, 512);
    gemm_kernel<__bf16, __bf16><<<dim3(112, 16), blk, 0, stream>>>(
        kv_a_n, wkv_b, kv, 2048, 14336, 512);
    // rope + K assembly
    build_qk_kernel<<<2048, 256, 0, stream>>>(latent, positions, q, kv, Kb);
    // causal flash attention
    mla_attn_kernel<<<dim3(32, 32), dim3(128), 0, stream>>>(q, Kb, kv, attn_o);
    // output projection
    gemm_kernel<__bf16, float><<<dim3(32, 16), blk, 0, stream>>>(
        attn_o, wo, out, 2048, 4096, 8192);
}
